// Encoder_65833258713548
// MI455X (gfx1250) — matrix operand order repaired and checked by reading, not compiled, not run on hardware
//
#include <hip/hip_runtime.h>
#include <hip/hip_bf16.h>

#define DEVINL __device__ __forceinline__

typedef __attribute__((ext_vector_type(16))) __bf16 v16bf;
typedef __attribute__((ext_vector_type(8)))  __bf16 v8bf;
typedef __attribute__((ext_vector_type(8)))  float  v8f;

static constexpr int kB  = 16;
static constexpr int kNG = 512;
static constexpr int kN  = kB * kNG;      // 8192
static constexpr int kE  = kN * 16;       // 131072
static constexpr int kL  = 64;
static constexpr int kH  = 256;
static constexpr float kSD = 0.0625f;     // 1/sqrt(256)

// ---------------------------------------------------------------------------
// WMMA fragment helpers (contiguous k order k = 16h + s, wave32)
// ---------------------------------------------------------------------------
DEVINL v16bf load_a_bf16(const __bf16* base, int ld) {
  // A operand, 16x32 (MxK): lanes 0-15 row M=lane, K 0..15;
  // lanes 16-31 row M=lane-16, K 16..31
  const int lane = threadIdx.x & 31;
  const int row  = lane & 15;
  const int k0   = (lane & 16) ? 16 : 0;
  const __bf16* p = base + (size_t)row * ld + k0;
  v8bf lo = *(const v8bf*)(p);
  v8bf hi = *(const v8bf*)(p + 8);
  v16bf f;
#pragma unroll
  for (int i = 0; i < 8; ++i) { f[i] = lo[i]; f[i + 8] = hi[i]; }
  return f;
}

DEVINL v16bf load_b_rows(const __bf16* Wt, int ld) {
  // B operand (KxN) built from rows of W (B = W_tile^T), W row-major [n][k]:
  // lanes 0-15 hold K=0..15 of column n=lane; lanes 16-31 hold K=16..31
  const int lane = threadIdx.x & 31;
  const int n  = lane & 15;
  const int kb = (lane & 16) ? 16 : 0;
  const __bf16* p = Wt + (size_t)n * ld + kb;
  v8bf lo = *(const v8bf*)(p);
  v8bf hi = *(const v8bf*)(p + 8);
  v16bf f;
#pragma unroll
  for (int i = 0; i < 8; ++i) { f[i] = lo[i]; f[i + 8] = hi[i]; }
  return f;
}

DEVINL v16bf load_b_tr(const __bf16* base, int ldk) {
  // B operand from naturally-oriented KxN memory: element (k,n)=base[k*ldk+n]
  const int lane = threadIdx.x & 31;
  const int n  = lane & 15;
  const int kb = (lane & 16) ? 16 : 0;
  v16bf f;
#pragma unroll
  for (int i = 0; i < 16; ++i) f[i] = base[(size_t)(kb + i) * ldk + n];
  return f;
}

DEVINL v8f wmma_bf16(v16bf a, v16bf b, v8f c) {
  return __builtin_amdgcn_wmma_f32_16x16x32_bf16(false, a, false, b, (short)0, c,
                                                 false, false);
}

DEVINL float sigf(float x) { return 1.0f / (1.0f + __expf(-x)); }

// ---------------------------------------------------------------------------
// Generic bf16 WMMA GEMM: C[M,Ncols] = act(alpha * A@W^T + bias)
// A row-major [M,K] bf16, W row-major [Ncols,K] bf16. One wave = 16x16 tile.
// ---------------------------------------------------------------------------
__global__ void __launch_bounds__(32)
gemm_bf16_kernel(const __bf16* __restrict__ A, const __bf16* __restrict__ W,
                 const float* __restrict__ bias,
                 float* __restrict__ C32, __bf16* __restrict__ C16,
                 int M, int Ncols, int K,
                 long long strideA, long long strideW, long long strideC,
                 float alpha, int act) {
  const int n0 = blockIdx.x * 16;
  const int m0 = blockIdx.y * 16;
  const long long bz = blockIdx.z;
  const __bf16* Ab = A + bz * strideA + (size_t)m0 * K;
  const __bf16* Wb = W + bz * strideW + (size_t)n0 * K;
  v8f acc = {0.f, 0.f, 0.f, 0.f, 0.f, 0.f, 0.f, 0.f};
  for (int k0 = 0; k0 < K; k0 += 32) {
    v16bf a = load_a_bf16(Ab + k0, K);
    v16bf b = load_b_rows(Wb + k0, K);
    acc = wmma_bf16(a, b, acc);
  }
  const int lane = threadIdx.x & 31;
  const int nloc = lane & 15;
  const int mb   = (lane & 16) ? 8 : 0;
  const int col  = n0 + nloc;
  const float bv = bias ? bias[col] : 0.0f;
#pragma unroll
  for (int v = 0; v < 8; ++v) {
    const int row = m0 + mb + v;
    if (row < M) {
      float x = acc[v] * alpha + bv;
      if (act == 1) x = fmaxf(x, 0.0f);
      else if (act == 2) x = sigf(x);
      const size_t idx = (size_t)bz * strideC + (size_t)row * Ncols + col;
      if (C32) C32[idx] = x;
      if (C16) C16[idx] = (__bf16)x;
    }
  }
}

// ---------------------------------------------------------------------------
// Fused comp kernel:  comp = H_c + sim[b(row)] * ( exp(Pr @ Qr^T) @ Hc_ )
// One wave handles 16 rows, streams all 8192 columns (flash-style, no NxN
// intermediate).  Pr row fragments stay resident in VGPRs.
// ---------------------------------------------------------------------------
__global__ void __launch_bounds__(32)
comp_fused_kernel(const __bf16* __restrict__ PR, const __bf16* __restrict__ QR,
                  const __bf16* __restrict__ HV16, const float* __restrict__ HC32,
                  const float* __restrict__ sim, const int* __restrict__ bvec,
                  float* __restrict__ C32, __bf16* __restrict__ C16) {
  __shared__ __bf16 eL[16 * 32];
  const int r0 = blockIdx.x * 16;
  const int lane = threadIdx.x & 31;
  const int nloc = lane & 15;
  const int mb   = (lane & 16) ? 8 : 0;
  const v8f vzero = {0.f, 0.f, 0.f, 0.f, 0.f, 0.f, 0.f, 0.f};

  v16bf aP[8];
#pragma unroll
  for (int kk = 0; kk < 8; ++kk)
    aP[kk] = load_a_bf16(PR + (size_t)r0 * kH + kk * 32, kH);

  v8f acc[16];
#pragma unroll
  for (int t = 0; t < 16; ++t) acc[t] = vzero;

  for (int j = 0; j < kN; j += 32) {
    if (j + 32 < kN) {
      __builtin_prefetch(QR + (size_t)(j + 32) * kH, 0, 1);
      __builtin_prefetch(HV16 + (size_t)(j + 32) * kH, 0, 1);
    }
    v8f s0 = vzero, s1 = vzero;
#pragma unroll
    for (int kk = 0; kk < 8; ++kk) {
      v16bf b0 = load_b_rows(QR + (size_t)j * kH + kk * 32, kH);
      v16bf b1 = load_b_rows(QR + (size_t)(j + 16) * kH + kk * 32, kH);
      s0 = wmma_bf16(aP[kk], b0, s0);
      s1 = wmma_bf16(aP[kk], b1, s1);
    }
#pragma unroll
    for (int v = 0; v < 8; ++v) {
      eL[(mb + v) * 32 + nloc]      = (__bf16)__expf(s0[v]);
      eL[(mb + v) * 32 + 16 + nloc] = (__bf16)__expf(s1[v]);
    }
    __syncthreads();
    v16bf ea = load_a_bf16(eL, 32);
#pragma unroll
    for (int ct = 0; ct < 16; ++ct) {
      v16bf bt = load_b_tr(HV16 + (size_t)j * kH + ct * 16, kH);
      acc[ct] = wmma_bf16(ea, bt, acc[ct]);
    }
    __syncthreads();
  }
  float srow[8];
#pragma unroll
  for (int v = 0; v < 8; ++v) srow[v] = sim[bvec[r0 + mb + v]];
#pragma unroll
  for (int ct = 0; ct < 16; ++ct) {
    const int col = ct * 16 + nloc;
#pragma unroll
    for (int v = 0; v < 8; ++v) {
      const int row = r0 + mb + v;
      const float val = HC32[(size_t)row * kH + col] + srow[v] * acc[ct][v];
      C32[(size_t)row * kH + col] = val;
      C16[(size_t)row * kH + col] = (__bf16)val;
    }
  }
}

// ---------------------------------------------------------------------------
// Bi-LSTM, scan over leading axis (16 steps), 64 parallel rows, WMMA gates.
// grid = 2 blocks (dir), 512 threads = 16 waves. Each wave: mt=wave&3 row
// tile, two u-tiles j; its 4 accumulators are the i/f/g/o gates for the same
// (l,u) -> no gate staging needed.
// ---------------------------------------------------------------------------
__global__ void __launch_bounds__(512)
lstm_kernel(const __bf16* __restrict__ XSEQ,
            const __bf16* __restrict__ WihF, const __bf16* __restrict__ WhhF,
            const float* __restrict__ bihF, const float* __restrict__ bhhF,
            const __bf16* __restrict__ WihB, const __bf16* __restrict__ WhhB,
            const float* __restrict__ bihB, const float* __restrict__ bhhB,
            float* __restrict__ OUT) {
  __shared__ float  c_s[64 * 128];
  __shared__ __bf16 h_s[64 * 128];
  const int dir = blockIdx.x;
  const __bf16* Wih = dir ? WihB : WihF;
  const __bf16* Whh = dir ? WhhB : WhhF;
  const float*  bih = dir ? bihB : bihF;
  const float*  bhh = dir ? bhhB : bhhF;
  for (int i = threadIdx.x; i < 64 * 128; i += 512) {
    c_s[i] = 0.0f;
    h_s[i] = (__bf16)0.0f;
  }
  __syncthreads();
  const int wave  = threadIdx.x >> 5;
  const int mt    = wave & 3;
  const int jbase = (wave >> 2) * 2;
  const int lane  = threadIdx.x & 31;
  const int nloc  = lane & 15;
  const int mb    = (lane & 16) ? 8 : 0;
  const v8f vzero = {0.f, 0.f, 0.f, 0.f, 0.f, 0.f, 0.f, 0.f};

  for (int s = 0; s < 16; ++s) {
    const int sx = dir ? (15 - s) : s;
    v8f acc[2][4];
#pragma unroll
    for (int t = 0; t < 2; ++t)
#pragma unroll
      for (int q = 0; q < 4; ++q) acc[t][q] = vzero;

    const __bf16* xb = XSEQ + ((size_t)sx * 64 + mt * 16) * kH;
    for (int kk = 0; kk < 8; ++kk) {
      v16bf a = load_a_bf16(xb + kk * 32, kH);
#pragma unroll
      for (int t = 0; t < 2; ++t) {
        const int j = jbase + t;
#pragma unroll
        for (int q = 0; q < 4; ++q) {
          v16bf b = load_b_rows(Wih + (size_t)(q * 128 + j * 16) * kH + kk * 32, kH);
          acc[t][q] = wmma_bf16(a, b, acc[t][q]);
        }
      }
    }
    for (int kk = 0; kk < 4; ++kk) {
      v16bf a = load_a_bf16(h_s + mt * 16 * 128 + kk * 32, 128);
#pragma unroll
      for (int t = 0; t < 2; ++t) {
        const int j = jbase + t;
#pragma unroll
        for (int q = 0; q < 4; ++q) {
          v16bf b = load_b_rows(Whh + (size_t)(q * 128 + j * 16) * 128 + kk * 32, 128);
          acc[t][q] = wmma_bf16(a, b, acc[t][q]);
        }
      }
    }
    __syncthreads();  // all h_s reads of this step finished
    const int sout = dir ? (15 - s) : s;
#pragma unroll
    for (int t = 0; t < 2; ++t) {
      const int u = (jbase + t) * 16 + nloc;
#pragma unroll
      for (int v = 0; v < 8; ++v) {
        const int l = mt * 16 + mb + v;
        const float gi = acc[t][0][v] + bih[u] + bhh[u];
        const float gf = acc[t][1][v] + bih[128 + u] + bhh[128 + u];
        const float gg = acc[t][2][v] + bih[256 + u] + bhh[256 + u];
        const float go = acc[t][3][v] + bih[384 + u] + bhh[384 + u];
        const float cn = sigf(gf) * c_s[l * 128 + u] + sigf(gi) * tanhf(gg);
        const float hh = sigf(go) * tanhf(cn);
        c_s[l * 128 + u] = cn;
        h_s[l * 128 + u] = (__bf16)hh;
        OUT[((size_t)sout * 64 + l) * 256 + dir * 128 + u] = hh;
      }
    }
    __syncthreads();
  }
}

// ---------------------------------------------------------------------------
// Small service kernels
// ---------------------------------------------------------------------------
__global__ void gather_rows_kernel(const float* __restrict__ emb,
                                   const int* __restrict__ idx,
                                   float* __restrict__ d32, __bf16* __restrict__ d16,
                                   int cols) {
  const int r = blockIdx.x;
  const int h = threadIdx.x;
  const float v = emb[(size_t)idx[r] * cols + h];
  if (d32) d32[(size_t)r * cols + h] = v;
  if (d16) d16[(size_t)r * cols + h] = (__bf16)v;
}

__global__ void f2b_kernel(const float* __restrict__ src, __bf16* __restrict__ dst, int n) {
  const int i = blockIdx.x * 256 + threadIdx.x;
  if (i < n) dst[i] = (__bf16)src[i];
}

__global__ void fillf_kernel(float* __restrict__ p, float v, int n) {
  const int i = blockIdx.x * 256 + threadIdx.x;
  if (i < n) p[i] = v;
}

__global__ void erf_kernel(const float* __restrict__ edge_emb,
                           const float* __restrict__ WRd, const float* __restrict__ WFd,
                           float* __restrict__ eR2, float* __restrict__ eF2) {
  const int t = threadIdx.x;  // 512 threads
  const int a = t >> 8, h = t & 255;
  float sr = 0.f, sf = 0.f;
  for (int k = 0; k < 256; ++k) {
    const float x = edge_emb[a * 256 + k];
    sr += x * WRd[h * 256 + k];
    sf += x * WFd[h * 256 + k];
  }
  eR2[a * 256 + h] = fmaxf(sr, 0.f);
  eF2[a * 256 + h] = sf;
}

__global__ void hop_scatter_kernel(const float* __restrict__ src, float* __restrict__ dst,
                                   const int* __restrict__ edges) {
  const int e = blockIdx.x;
  const int h = threadIdx.x;
  const int r = edges[e];
  const int c = edges[kE + e];
  unsafeAtomicAdd(&dst[(size_t)c * kH + h], src[(size_t)r * kH + h]);
}

__global__ void edge_qe_kernel(const float* __restrict__ Q, const float* __restrict__ eR2,
                               const int* __restrict__ edges, const int* __restrict__ attr,
                               float* __restrict__ ABLK) {
  const int e = blockIdx.x * 256 + threadIdx.x;
  if (e >= kE) return;
  const int r = edges[e];
  const int c = edges[kE + e];
  if ((r >> 9) != (c >> 9)) return;  // cross-segment edges are masked out
  const float* q  = Q + (size_t)r * kH;
  const float* er = eR2 + (size_t)attr[e] * kH;
  float s = 0.f;
  for (int i = 0; i < kH; i += 4)
    s += q[i] * er[i] + q[i + 1] * er[i + 1] + q[i + 2] * er[i + 2] + q[i + 3] * er[i + 3];
  unsafeAtomicAdd(&ABLK[(size_t)r * 512 + (c & 511)], s * kSD);
}

__global__ void __launch_bounds__(32)
softmax_rows_kernel(float* __restrict__ Ab) {
  float* p = Ab + (size_t)blockIdx.x * 512;
  const int lane = threadIdx.x;
  float vals[16];
  float m = -3.4e38f;
#pragma unroll
  for (int i = 0; i < 16; ++i) { vals[i] = p[lane + i * 32]; m = fmaxf(m, vals[i]); }
#pragma unroll
  for (int s = 16; s >= 1; s >>= 1) m = fmaxf(m, __shfl_xor(m, s, 32));
  float sum = 0.f;
#pragma unroll
  for (int i = 0; i < 16; ++i) { vals[i] = __expf(vals[i] - m); sum += vals[i]; }
#pragma unroll
  for (int s = 16; s >= 1; s >>= 1) sum += __shfl_xor(sum, s, 32);
  const float inv = 1.0f / sum;
#pragma unroll
  for (int i = 0; i < 16; ++i) p[lane + i * 32] = vals[i] * inv;
}

__global__ void edge_msg_kernel(const float* __restrict__ ABLK, const float* __restrict__ V,
                                const float* __restrict__ eF2,
                                const int* __restrict__ edges, const int* __restrict__ attr,
                                float* __restrict__ dst) {
  const int e = blockIdx.x;
  const int h = threadIdx.x;
  const int r = edges[e];
  const int c = edges[kE + e];
  if ((r >> 9) != (c >> 9)) return;
  const float a = ABLK[(size_t)r * 512 + (c & 511)];
  const float m = a * (V[(size_t)r * kH + h] + eF2[(size_t)attr[e] * kH + h]);
  unsafeAtomicAdd(&dst[(size_t)c * kH + h], m);
}

__global__ void zcat_kernel(const float* __restrict__ hv, const float* __restrict__ hvd,
                            __bf16* __restrict__ zc) {
  const int n = blockIdx.x;
  const int h = threadIdx.x;
  const float a = hv[(size_t)n * kH + h];
  const float b = hvd[(size_t)n * kH + h];
  __bf16* p = zc + (size_t)n * 1024;
  p[h] = (__bf16)a; p[256 + h] = (__bf16)b;
  p[512 + h] = (__bf16)(a - b); p[768 + h] = (__bf16)(a * b);
}

__global__ void fuse_kernel(const float* __restrict__ Z, const float* __restrict__ hv,
                            const float* __restrict__ hvd,
                            float* __restrict__ F32, __bf16* __restrict__ F16, int n) {
  const int i = blockIdx.x * 256 + threadIdx.x;
  if (i >= n) return;
  const float z = Z[i];
  const float f = z * hv[i] + (1.0f - z) * hvd[i];
  F32[i] = f; F16[i] = (__bf16)f;
}

__global__ void gru_elem_kernel(const float* __restrict__ GI, const float* __restrict__ GH,
                                const float* __restrict__ FUSE, float* __restrict__ out) {
  const int i = blockIdx.x * 256 + threadIdx.x;
  if (i >= kN * kH) return;
  const int n = i >> 8, u = i & 255;
  const float* gi = GI + (size_t)n * 768;
  const float* gh = GH + (size_t)n * 768;
  const float r  = sigf(gi[u] + gh[u]);
  const float zz = sigf(gi[256 + u] + gh[256 + u]);
  const float nn = tanhf(gi[512 + u] + r * gh[512 + u]);
  const float f  = (1.0f - zz) * nn + zz * FUSE[i];
  const int b = n >> 9, ii = n & 511;
  out[((size_t)b * 576 + 64 + ii) * 256 + u] = f;   // enc_outs NG part
}

__global__ void h_out_kernel(const float* __restrict__ OUT, const float* __restrict__ sim,
                             float* __restrict__ out) {
  const int i = blockIdx.x * 256 + threadIdx.x;
  if (i >= kB * kL * kH) return;
  const int b = i >> 14;
  const int t = (i >> 8) & 63;
  const int u = i & 255;
  out[((size_t)b * 576 + t) * 256 + u] = sim[b] * OUT[i];
}

__global__ void segmax_kernel(const float* __restrict__ out, float* __restrict__ GR) {
  const int b = blockIdx.x;
  const int h = threadIdx.x;
  float m = -3.4e38f;
  const float* p = out + ((size_t)b * 576 + 64) * 256 + h;
  for (int i = 0; i < 512; ++i) m = fmaxf(m, p[(size_t)i * 256]);
  GR[b * 256 + h] = m;
}

__global__ void zc2_kernel(const float* __restrict__ GR, const float* __restrict__ OUT,
                           __bf16* __restrict__ zc2) {
  const int b = blockIdx.x;
  const int h = threadIdx.x;
  const float g  = GR[b * 256 + h];
  const float hn = OUT[((size_t)b * 64 + 63) * 256 + h];
  __bf16* p = zc2 + (size_t)b * 1024;
  p[h] = (__bf16)g; p[256 + h] = (__bf16)hn;
  p[512 + h] = (__bf16)(g * hn); p[768 + h] = (__bf16)(g - hn);
}

__global__ void decinit_kernel(const float* __restrict__ Z2, const float* __restrict__ GR,
                               const float* __restrict__ OUT, float* __restrict__ dst) {
  const int i = blockIdx.x * 256 + threadIdx.x;
  if (i >= kB * kH) return;
  const int b = i >> 8, h = i & 255;
  const float hn = OUT[((size_t)b * 64 + 63) * 256 + h];
  const float z = Z2[i];
  dst[i] = z * GR[i] + (1.0f - z) * hn;
}

// ---------------------------------------------------------------------------
// Host orchestration
// ---------------------------------------------------------------------------
extern "C" void kernel_launch(void* const* d_in, const int* in_sizes, int n_in,
                              void* d_out, int out_size, void* d_ws, size_t ws_size,
                              hipStream_t stream) {
  (void)in_sizes; (void)n_in; (void)out_size; (void)ws_size;
  const int*   src_node  = (const int*)d_in[0];
  const int*   src_edge  = (const int*)d_in[1];
  const int*   src_attr  = (const int*)d_in[2];
  const int*   rev_node  = (const int*)d_in[3];
  const int*   rev_com   = (const int*)d_in[4];
  const int*   batch_vec = (const int*)d_in[6];
  const float* sim       = (const float*)d_in[7];
  const float* code_emb  = (const float*)d_in[8];
  const float* com_emb   = (const float*)d_in[9];
  const float* edge_emb  = (const float*)d_in[10];
  const float* W_C    = (const float*)d_in[11];
  const float* W_Q    = (const float*)d_in[12];
  const float* Wz_enc = (const float*)d_in[13];
  const float* bz_enc = (const float*)d_in[14];
  const float* lWihF  = (const float*)d_in[15];
  const float* lWhhF  = (const float*)d_in[16];
  const float* lbihF  = (const float*)d_in[17];
  const float* lbhhF  = (const float*)d_in[18];
  const float* lWihB  = (const float*)d_in[19];
  const float* lWhhB  = (const float*)d_in[20];
  const float* lbihB  = (const float*)d_in[21];
  const float* lbhhB  = (const float*)d_in[22];
  const float* WQd    = (const float*)d_in[23];
  const float* WKd    = (const float*)d_in[24];
  const float* WRd    = (const float*)d_in[25];
  const float* WVd    = (const float*)d_in[26];
  const float* WFd    = (const float*)d_in[27];
  const float* gWih   = (const float*)d_in[28];
  const float* gWhh   = (const float*)d_in[29];
  const float* gbih   = (const float*)d_in[30];
  const float* gbhh   = (const float*)d_in[31];
  const float* Wz_h   = (const float*)d_in[32];
  const float* bz_h   = (const float*)d_in[33];
  float* out = (float*)d_out;

  char* base = (char*)d_ws;
  size_t off = 0;
  auto alloc = [&](size_t bytes) -> char* {
    char* p = base + off;
    off = (off + bytes + 255) & ~(size_t)255;
    return p;
  };
  const size_t F32NH = (size_t)kN * kH * 4;
  const size_t B16NH = (size_t)kN * kH * 2;

  __bf16* WC16   = (__bf16*)alloc(65536 * 2);
  __bf16* WQ16   = (__bf16*)alloc(65536 * 2);
  __bf16* WQd16  = (__bf16*)alloc(65536 * 2);
  __bf16* WKd16  = (__bf16*)alloc(65536 * 2);
  __bf16* WVd16  = (__bf16*)alloc(65536 * 2);
  __bf16* WzH16  = (__bf16*)alloc(262144 * 2);
  __bf16* WzE16  = (__bf16*)alloc(262144 * 2);
  __bf16* GWih16 = (__bf16*)alloc(196608 * 2);
  __bf16* GWhh16 = (__bf16*)alloc(196608 * 2);
  __bf16* LWihF16 = (__bf16*)alloc(131072 * 2);
  __bf16* LWhhF16 = (__bf16*)alloc(65536 * 2);
  __bf16* LWihB16 = (__bf16*)alloc(131072 * 2);
  __bf16* LWhhB16 = (__bf16*)alloc(65536 * 2);
  float*  ER2   = (float*)alloc(2 * kH * 4);
  float*  EF2   = (float*)alloc(2 * kH * 4);
  __bf16* XSEQ16 = (__bf16*)alloc((size_t)kB * kL * kH * 2);
  float*  OUT32  = (float*)alloc((size_t)kB * kL * 256 * 4);
  float*  GR     = (float*)alloc(kB * kH * 4);
  float*  Z2     = (float*)alloc(kB * kH * 4);
  __bf16* ZC2    = (__bf16*)alloc(kB * 1024 * 2);
  float*  HC32   = (float*)alloc(F32NH);
  __bf16* HC16   = (__bf16*)alloc(B16NH);
  __bf16* HCr16  = (__bf16*)alloc(B16NH);
  __bf16* PR16   = (__bf16*)alloc(B16NH);   // reused as Q16 during hops
  __bf16* QR16   = (__bf16*)alloc(B16NH);   // reused as KB16 during hops
  float*  COMP32 = (float*)alloc(F32NH);
  __bf16* COMP16 = (__bf16*)alloc(B16NH);
  __bf16* HVD16  = (__bf16*)alloc(B16NH);
  float*  Q32    = (float*)alloc(F32NH);
  float*  V32    = (float*)alloc(F32NH);
  float*  HVA    = (float*)alloc(F32NH);    // h_v final; + next 2 = GH alias
  float*  HVB    = (float*)alloc(F32NH);
  float*  HVDA   = (float*)alloc(F32NH);    // h_v_ final
  float*  HVDB   = (float*)alloc(F32NH);
  float*  ABLK   = (float*)alloc((size_t)16 * 512 * 512 * 4);  // + ZCAT = GI alias
  __bf16* ZCAT16 = (__bf16*)alloc((size_t)kN * 1024 * 2);
  float*  Z32    = (float*)alloc(F32NH);
  float*  FUSE32 = (float*)alloc(F32NH);
  __bf16* FUSE16 = (__bf16*)alloc(B16NH);
  float* GI32 = ABLK;   // dead by GRU time (25.2MB into 33.5MB region)
  float* GH32 = HVA;    // dead by GRU time (25.2MB over HVA/HVB/HVDA)

  auto f2b = [&](const float* s, __bf16* d, int n) {
    f2b_kernel<<<(n + 255) / 256, 256, 0, stream>>>(s, d, n);
  };
  auto gemm = [&](const __bf16* A, const __bf16* W, const float* bias, float* C32,
                  __bf16* C16, int M, int Ncols, int K, long long sA, long long sW,
                  long long sC, int nz, float alpha, int act) {
    dim3 g(Ncols / 16, (M + 15) / 16, nz);
    gemm_bf16_kernel<<<g, 32, 0, stream>>>(A, W, bias, C32, C16, M, Ncols, K, sA, sW,
                                           sC, alpha, act);
  };

  // --- weight conversions ---------------------------------------------------
  f2b(W_C, WC16, 65536);     f2b(W_Q, WQ16, 65536);
  f2b(WQd, WQd16, 65536);    f2b(WKd, WKd16, 65536);   f2b(WVd, WVd16, 65536);
  f2b(Wz_h, WzH16, 262144);  f2b(Wz_enc, WzE16, 262144);
  f2b(gWih, GWih16, 196608); f2b(gWhh, GWhh16, 196608);
  f2b(lWihF, LWihF16, 131072); f2b(lWhhF, LWhhF16, 65536);
  f2b(lWihB, LWihB16, 131072); f2b(lWhhB, LWhhB16, 65536);

  // --- gathers --------------------------------------------------------------
  gather_rows_kernel<<<kN, kH, 0, stream>>>(code_emb, src_node, HC32, HC16, kH);
  gather_rows_kernel<<<kN, kH, 0, stream>>>(code_emb, rev_node, nullptr, HCr16, kH);
  gather_rows_kernel<<<kB * kL, kH, 0, stream>>>(com_emb, rev_com, nullptr, XSEQ16, kH);

  // --- BiLSTM (scan over leading/batch axis, 64 parallel rows) --------------
  lstm_kernel<<<2, 512, 0, stream>>>(XSEQ16, LWihF16, LWhhF16, lbihF, lbhhF,
                                     LWihB16, LWhhB16, lbihB, lbhhB, OUT32);

  // --- Pr / Qr projections, then fused exp-attention comp -------------------
  gemm(HC16, WC16, nullptr, nullptr, PR16, kN, kH, kH, 0, 0, 0, 1, 1.f, 1);
  gemm(HCr16, WQ16, nullptr, nullptr, QR16, kN, kH, kH, 0, 0, 0, 1, 1.f, 1);
  comp_fused_kernel<<<kN / 16, 32, 0, stream>>>(PR16, QR16, HCr16, HC32, sim,
                                                batch_vec, COMP32, COMP16);

  // --- edge feature projections ---------------------------------------------
  erf_kernel<<<1, 512, 0, stream>>>(edge_emb, WRd, WFd, ER2, EF2);

  // --- plain graph propagation: h_v (2 hops, identity + scatter-add) --------
  const size_t nhBytes = F32NH;
  hipMemcpyAsync(HVB, COMP32, nhBytes, hipMemcpyDeviceToDevice, stream);
  hop_scatter_kernel<<<kE, kH, 0, stream>>>(COMP32, HVB, src_edge);
  hipMemcpyAsync(HVA, HVB, nhBytes, hipMemcpyDeviceToDevice, stream);
  hop_scatter_kernel<<<kE, kH, 0, stream>>>(HVB, HVA, src_edge);

  // --- dynamic attention hops: h_v_ ------------------------------------------
  __bf16* Q16  = PR16;
  __bf16* KB16 = QR16;
  for (int hop = 0; hop < 2; ++hop) {
    const __bf16* srcb = (hop == 0) ? COMP16 : HVD16;
    float* dst = (hop == 0) ? HVDB : HVDA;
    if (hop == 1) f2b(HVDB, HVD16, kN * kH);
    gemm(srcb, WQd16, nullptr, Q32, Q16, kN, kH, kH, 0, 0, 0, 1, 1.f, 1);
    gemm(srcb, WKd16, nullptr, nullptr, KB16, kN, kH, kH, 0, 0, 0, 1, 1.f, 1);
    gemm(srcb, WVd16, nullptr, V32, nullptr, kN, kH, kH, 0, 0, 0, 1, 1.f, 0);
    // block-diagonal S = q @ kb^T * sd   (16 batched 512x512x256 GEMMs)
    gemm(Q16, KB16, nullptr, ABLK, nullptr, kNG, kNG, kH,
         (long long)kNG * kH, (long long)kNG * kH, (long long)kNG * kNG, kB, kSD, 0);
    edge_qe_kernel<<<(kE + 255) / 256, 256, 0, stream>>>(Q32, ER2, src_edge, src_attr, ABLK);
    softmax_rows_kernel<<<kN, 32, 0, stream>>>(ABLK);
    fillf_kernel<<<(kN * kH + 255) / 256, 256, 0, stream>>>(dst, 0.f, kN * kH);
    edge_msg_kernel<<<kE, kH, 0, stream>>>(ABLK, V32, EF2, src_edge, src_attr, dst);
  }

  // --- gated fusion z / fuse --------------------------------------------------
  zcat_kernel<<<kN, kH, 0, stream>>>(HVA, HVDA, ZCAT16);
  gemm(ZCAT16, WzH16, bz_h, Z32, nullptr, kN, kH, 1024, 0, 0, 0, 1, 1.f, 2);
  fuse_kernel<<<(kN * kH + 255) / 256, 256, 0, stream>>>(Z32, HVA, HVDA, FUSE32,
                                                         FUSE16, kN * kH);

  // --- GRU cell ---------------------------------------------------------------
  gemm(COMP16, GWih16, gbih, GI32, nullptr, kN, 768, kH, 0, 0, 0, 1, 1.f, 0);
  gemm(FUSE16, GWhh16, gbhh, GH32, nullptr, kN, 768, kH, 0, 0, 0, 1, 1.f, 0);
  gru_elem_kernel<<<(kN * kH + 255) / 256, 256, 0, stream>>>(GI32, GH32, FUSE32, out);

  // --- outputs ----------------------------------------------------------------
  h_out_kernel<<<(kB * kL * kH + 255) / 256, 256, 0, stream>>>(OUT32, sim, out);
  segmax_kernel<<<kB, kH, 0, stream>>>(out, GR);
  zc2_kernel<<<kB, kH, 0, stream>>>(GR, OUT32, ZC2);
  gemm(ZC2, WzE16, bz_enc, Z2, nullptr, kB, kH, 1024, 0, 0, 0, 1, 1.f, 2);
  decinit_kernel<<<(kB * kH + 255) / 256, 256, 0, stream>>>(Z2, GR, OUT32,
                                                            out + (size_t)kB * 576 * kH);
}
